// _PairwiseMessages_73607149519580
// MI455X (gfx1250) — matrix operand order repaired and checked by reading, not compiled, not run on hardware
//
#include <hip/hip_runtime.h>

typedef __attribute__((ext_vector_type(16))) _Float16 v16h;
typedef __attribute__((ext_vector_type(8)))  _Float16 v8h;
typedef __attribute__((ext_vector_type(8)))  float    v8f;

#define B_   2
#define NQ_  512
#define NK_  512
#define DEQ  32
#define DMSG 64
#define DFF  128
#define DOUT 64
#define DQI  128
#define DKI  128

// workspace layout (dword offsets)
#define WS_WP   0                              // 8192 dwords: packed f16 W1-small (16 tiles) + W2 (16 tiles)
#define WS_A    8192                           // B*NQ*128 f32  (q_msg @ W1[0:64])
#define WS_BK   (WS_A  + B_*NQ_*DFF)           // B*NK*128 f32  (k_msg @ W1[64:128] + b1)
#define WS_QN   (WS_BK + B_*NK_*DFF)           // B*NQ*32  f32
#define WS_KN   (WS_QN + B_*NQ_*DEQ)           // B*NK*32  f32

__device__ __forceinline__ float fast_silu(float x) {
    // x * sigmoid(x) with raw HW transcendentals (v_exp_f32 + v_rcp_f32)
    float t = __builtin_amdgcn_exp2f(x * -1.44269504f);
    return x * __builtin_amdgcn_rcpf(1.0f + t);
}

// ---------------------------------------------------------------------------
// Pack W1[128:192,:] (64x128) and W2 (128x64) into f16 WMMA B-operand tiles.
// B 32x16 f16 layout: lane L -> n = L%16, g = L/16; dword v holds tile rows
// W1: K = 8g+2v (v<4) / 16+8g+2(v-4) (v>=4) (lo), K+1 (hi); W2 slot: 16g+2v.  Tile stored [lane][8 dwords].
// For W2 (GEMM2) the K dimension is permuted so the per-wave h-buffer can be
// written with contiguous b128 stores:  slot S = kt*32 + sub*16 + 8*gA + j
// holds logical feature n = j*16 + (kt*4 + 2*sub + gA).
// ---------------------------------------------------------------------------
__global__ void pack_weights(const float* __restrict__ W1,
                             const float* __restrict__ W2,
                             unsigned int* __restrict__ wp) {
    int t = blockIdx.x * blockDim.x + threadIdx.x;   // 0..8191
    if (t >= 8192) return;
    bool isW2 = t >= 4096;
    int loc  = isW2 ? t - 4096 : t;
    int tile = loc >> 8;          // /256 dwords per tile
    int lane = (loc >> 3) & 31;
    int v    = loc & 7;
    int g = lane >> 4, n = lane & 15;
    float f0, f1;
    if (!isW2) {
        int kt = tile >> 3, nt = tile & 7;           // tile = kt*8 + nt
        int K0 = kt * 32 + (v < 4 ? 8 * g + 2 * v : 16 + 8 * g + 2 * (v - 4));   // feature row = 128 + K0
        int col = nt * 16 + n;
        f0 = W1[(128 + K0) * DFF + col];
        f1 = W1[(128 + K0 + 1) * DFF + col];
    } else {
        int kt = tile >> 2, nt = tile & 3;           // tile = kt*4 + nt
        int col = nt * 16 + n;
        // within-tile slot s = 16*g + 2*v + h ; h = 0,1
        int s0 = 2 * v;                               // low 4 bits of slot (h=0)
        int gA0 = (s0 >> 3) & 1,        j0 = s0 & 7;
        int gA1 = ((s0 + 1) >> 3) & 1,  j1 = (s0 + 1) & 7;
        int n0 = j0 * 16 + kt * 4 + 2 * g + gA0;      // sub = g (tile half)
        int n1 = j1 * 16 + kt * 4 + 2 * g + gA1;
        f0 = W2[n0 * DOUT + col];
        f1 = W2[n1 * DOUT + col];
    }
    union { _Float16 h[2]; unsigned int u; } pk;
    pk.h[0] = (_Float16)f0;
    pk.h[1] = (_Float16)f1;
    wp[t] = pk.u;
}

// ---------------------------------------------------------------------------
// Per-row precompute: msg = inv @ Wm + bm; outA = msg @ W1[w1off:w1off+64] (+badd);
// outN = sum_c equi[c]^2.   One block per row, 128 threads.
// ---------------------------------------------------------------------------
__global__ void row_prep(const float* __restrict__ inv, const float* __restrict__ equi,
                         const float* __restrict__ Wm, const float* __restrict__ bm,
                         const float* __restrict__ W1, int w1off, const float* badd,
                         float* __restrict__ outA, float* __restrict__ outN, int dinv) {
    int r = blockIdx.x;
    int tid = threadIdx.x;
    __shared__ float msg[DMSG];
    if (tid < DMSG) {
        float s = bm[tid];
        for (int j = 0; j < dinv; ++j) s += inv[(size_t)r * dinv + j] * Wm[j * DMSG + tid];
        msg[tid] = s;
    }
    __syncthreads();
    float a = badd ? badd[tid] : 0.0f;
    for (int j = 0; j < DMSG; ++j) a += msg[j] * W1[(w1off + j) * DFF + tid];
    outA[(size_t)r * DFF + tid] = a;
    if (tid < DEQ) {
        float x0 = equi[(size_t)r * 96 + tid];
        float x1 = equi[(size_t)r * 96 + 32 + tid];
        float x2 = equi[(size_t)r * 96 + 64 + tid];
        outN[(size_t)r * DEQ + tid] = x0 * x0 + x1 * x1 + x2 * x2;
    }
}

// ---------------------------------------------------------------------------
// Main pairwise kernel. 128 threads = 4 waves. Each wave: 16 q-rows x 8 k's.
// Per k: X = [dot(32)|dist(32)] in f16 A-layout,
//   h = silu(X @ W1s + A_q + B_k)   -> 16x v_wmma_f32_16x16x32_f16
//   out = h @ W2perm + b2           -> 16x v_wmma_f32_16x16x32_f16
// h transposed D-layout -> A-layout via per-wave LDS buffer with permuted
// columns: phys[m][c*8 + nt] = h[m][nt*16 + c]  (b128 stores AND b128 loads).
// Output written with non-temporal stores (write-once stream, 134 MB).
// ---------------------------------------------------------------------------
__launch_bounds__(128)
__global__ void pair_mlp(const float* __restrict__ q_equi, const float* __restrict__ k_equi,
                         const float* __restrict__ Aq, const float* __restrict__ Bk,
                         const float* __restrict__ qn, const float* __restrict__ kn,
                         const unsigned int* __restrict__ wp, const float* __restrict__ b2,
                         float* __restrict__ out) {
    __shared__ __align__(32) unsigned int wlds[8192];     // 32KB packed weights
    __shared__ __align__(32) _Float16 hbuf[4 * 16 * DFF]; // 16KB, 4KB per wave

    const int tid = threadIdx.x;
    #pragma unroll
    for (int i = 0; i < 64; ++i) wlds[tid + i * 128] = wp[tid + i * 128];
    __syncthreads();

    const int idx  = blockIdx.x;        // 0..1023
    const int kblk = idx & 15;          // 16 k-blocks of 32
    const int qblk = (idx >> 4) & 31;   // 32 q-blocks of 16
    const int b    = idx >> 9;

    const int wave = tid >> 5;
    const int lane = tid & 31;
    const int g    = lane >> 4;
    const int n16  = lane & 15;

    const int q0     = qblk * 16;
    const int kstart = kblk * 32 + wave * 8;

    // q-side registers: lane holds row m = n16, e-runs {8g..8g+7} and {16+8g..+7}
    const int qrow  = b * NQ_ + q0 + n16;
    const int eoff0 = 8 * g;        // first contiguous e-run
    const int eoff1 = 16 + 8 * g;   // second contiguous e-run
    float qe[3][16], qn_r[16];
    #pragma unroll
    for (int c = 0; c < 3; ++c) {
        float4 a0 = *(const float4*)&q_equi[(size_t)qrow * 96 + c * 32 + eoff0];
        float4 a1 = *(const float4*)&q_equi[(size_t)qrow * 96 + c * 32 + eoff0 + 4];
        float4 a2 = *(const float4*)&q_equi[(size_t)qrow * 96 + c * 32 + eoff1];
        float4 a3 = *(const float4*)&q_equi[(size_t)qrow * 96 + c * 32 + eoff1 + 4];
        qe[c][0]=a0.x; qe[c][1]=a0.y; qe[c][2]=a0.z; qe[c][3]=a0.w;
        qe[c][4]=a1.x; qe[c][5]=a1.y; qe[c][6]=a1.z; qe[c][7]=a1.w;
        qe[c][8]=a2.x; qe[c][9]=a2.y; qe[c][10]=a2.z; qe[c][11]=a2.w;
        qe[c][12]=a3.x; qe[c][13]=a3.y; qe[c][14]=a3.z; qe[c][15]=a3.w;
    }
    {
        float4 a0 = *(const float4*)&qn[(size_t)qrow * 32 + eoff0];
        float4 a1 = *(const float4*)&qn[(size_t)qrow * 32 + eoff0 + 4];
        float4 a2 = *(const float4*)&qn[(size_t)qrow * 32 + eoff1];
        float4 a3 = *(const float4*)&qn[(size_t)qrow * 32 + eoff1 + 4];
        qn_r[0]=a0.x; qn_r[1]=a0.y; qn_r[2]=a0.z; qn_r[3]=a0.w;
        qn_r[4]=a1.x; qn_r[5]=a1.y; qn_r[6]=a1.z; qn_r[7]=a1.w;
        qn_r[8]=a2.x; qn_r[9]=a2.y; qn_r[10]=a2.z; qn_r[11]=a2.w;
        qn_r[12]=a3.x; qn_r[13]=a3.y; qn_r[14]=a3.z; qn_r[15]=a3.w;
    }

    // A_q in D-layout: value for row m = 8*g + r, col n = nt*16 + n16
    float aq[8][8];
    #pragma unroll
    for (int nt = 0; nt < 8; ++nt)
        #pragma unroll
        for (int r = 0; r < 8; ++r)
            aq[nt][r] = Aq[(size_t)(b * NQ_ + q0 + 8 * g + r) * DFF + nt * 16 + n16];

    float b2r[4];
    #pragma unroll
    for (int nt = 0; nt < 4; ++nt) b2r[nt] = b2[nt * 16 + n16];

    _Float16* hb = &hbuf[wave * 16 * DFF];

    for (int i = 0; i < 8; ++i) {
        const int k    = kstart + i;
        const int krow = b * NK_ + k;

        // k-side vector loads
        float ke[3][16], kn_r[16];
        #pragma unroll
        for (int c = 0; c < 3; ++c) {
            float4 a0 = *(const float4*)&k_equi[(size_t)krow * 96 + c * 32 + eoff0];
            float4 a1 = *(const float4*)&k_equi[(size_t)krow * 96 + c * 32 + eoff0 + 4];
            float4 a2 = *(const float4*)&k_equi[(size_t)krow * 96 + c * 32 + eoff1];
            float4 a3 = *(const float4*)&k_equi[(size_t)krow * 96 + c * 32 + eoff1 + 4];
            ke[c][0]=a0.x; ke[c][1]=a0.y; ke[c][2]=a0.z; ke[c][3]=a0.w;
            ke[c][4]=a1.x; ke[c][5]=a1.y; ke[c][6]=a1.z; ke[c][7]=a1.w;
            ke[c][8]=a2.x; ke[c][9]=a2.y; ke[c][10]=a2.z; ke[c][11]=a2.w;
            ke[c][12]=a3.x; ke[c][13]=a3.y; ke[c][14]=a3.z; ke[c][15]=a3.w;
        }
        {
            float4 a0 = *(const float4*)&kn[(size_t)krow * 32 + eoff0];
            float4 a1 = *(const float4*)&kn[(size_t)krow * 32 + eoff0 + 4];
            float4 a2 = *(const float4*)&kn[(size_t)krow * 32 + eoff1];
            float4 a3 = *(const float4*)&kn[(size_t)krow * 32 + eoff1 + 4];
            kn_r[0]=a0.x; kn_r[1]=a0.y; kn_r[2]=a0.z; kn_r[3]=a0.w;
            kn_r[4]=a1.x; kn_r[5]=a1.y; kn_r[6]=a1.z; kn_r[7]=a1.w;
            kn_r[8]=a2.x; kn_r[9]=a2.y; kn_r[10]=a2.z; kn_r[11]=a2.w;
            kn_r[12]=a3.x; kn_r[13]=a3.y; kn_r[14]=a3.z; kn_r[15]=a3.w;
        }

        // dot / dist features (raw v_sqrt_f32)
        v16h X0, X1;
        #pragma unroll
        for (int j = 0; j < 16; ++j) {
            float d = qe[0][j] * ke[0][j] + qe[1][j] * ke[1][j] + qe[2][j] * ke[2][j];
            float s = qn_r[j] + kn_r[j] - 2.0f * d;
            X0[j] = (_Float16)d;
            X1[j] = (_Float16)__builtin_amdgcn_sqrtf(fmaxf(s, 0.0f));
        }

        float bkv[8];
        #pragma unroll
        for (int nt = 0; nt < 8; ++nt) bkv[nt] = Bk[(size_t)krow * DFF + nt * 16 + n16];

        // GEMM1: h = silu(X @ W1s + Aq + Bk)   (N = 128, K = 64)
        float hrow[8][8];   // [nt][r]
        #pragma unroll
        for (int nt = 0; nt < 8; ++nt) {
            v8f c;
            #pragma unroll
            for (int r = 0; r < 8; ++r) c[r] = aq[nt][r] + bkv[nt];
            v16h B0 = *(const v16h*)&wlds[(0 * 8 + nt) * 256 + lane * 8];
            c = __builtin_amdgcn_wmma_f32_16x16x32_f16(false, X0, false, B0, (short)0, c, false, false);
            v16h B1 = *(const v16h*)&wlds[(1 * 8 + nt) * 256 + lane * 8];
            c = __builtin_amdgcn_wmma_f32_16x16x32_f16(false, X1, false, B1, (short)0, c, false, false);
            #pragma unroll
            for (int r = 0; r < 8; ++r) hrow[nt][r] = fast_silu(c[r]);
        }
        // write h to LDS, permuted-contiguous: phys[m][n16*8 + nt], m = 8g + r
        #pragma unroll
        for (int r = 0; r < 8; ++r) {
            v8h w;
            #pragma unroll
            for (int nt = 0; nt < 8; ++nt) w[nt] = (_Float16)hrow[nt][r];
            *(v8h*)&hb[(8 * g + r) * DFF + n16 * 8] = w;   // ds_store_b128
        }

        // GEMM2: out = h @ W2perm + b2  (same-wave DS ordering: writes visible)
        v8f o[4];
        #pragma unroll
        for (int nt = 0; nt < 4; ++nt)
            #pragma unroll
            for (int r = 0; r < 8; ++r) o[nt][r] = b2r[nt];
        #pragma unroll
        for (int kt = 0; kt < 4; ++kt) {
            v8h lo = *(const v8h*)&hb[n16 * DFF + (kt * 4 + 2 * g) * 8];       // sub=g, gA=0
            v8h hi = *(const v8h*)&hb[n16 * DFF + (kt * 4 + 2 * g + 1) * 8];   // sub=g, gA=1
            v16h a = __builtin_shufflevector(lo, hi, 0,1,2,3,4,5,6,7,8,9,10,11,12,13,14,15);
            #pragma unroll
            for (int nt = 0; nt < 4; ++nt) {
                v16h Bt = *(const v16h*)&wlds[4096 + (kt * 4 + nt) * 256 + lane * 8];
                o[nt] = __builtin_amdgcn_wmma_f32_16x16x32_f16(false, a, false, Bt, (short)0, o[nt], false, false);
            }
        }

        // store 16x64 fp32 tile; non-temporal (write-once stream, never re-read)
        #pragma unroll
        for (int nt = 0; nt < 4; ++nt)
            #pragma unroll
            for (int r = 0; r < 8; ++r)
                __builtin_nontemporal_store(o[nt][r],
                    &out[((size_t)(b * NQ_ + q0 + 8 * g + r) * NK_ + k) * 64 + nt * 16 + n16]);
    }
}

extern "C" void kernel_launch(void* const* d_in, const int* in_sizes, int n_in,
                              void* d_out, int out_size, void* d_ws, size_t ws_size,
                              hipStream_t stream) {
    const float* q_equi = (const float*)d_in[0];
    const float* q_inv  = (const float*)d_in[1];
    const float* k_equi = (const float*)d_in[2];
    const float* k_inv  = (const float*)d_in[3];
    const float* Wq     = (const float*)d_in[4];
    const float* bq     = (const float*)d_in[5];
    const float* Wk     = (const float*)d_in[6];
    const float* bk     = (const float*)d_in[7];
    const float* W1     = (const float*)d_in[8];
    const float* b1     = (const float*)d_in[9];
    const float* W2     = (const float*)d_in[10];
    const float* b2     = (const float*)d_in[11];
    float* out = (float*)d_out;

    unsigned int* wsu = (unsigned int*)d_ws;
    float*        wsf = (float*)d_ws;
    unsigned int* wp  = wsu + WS_WP;
    float* A   = wsf + WS_A;
    float* Bk_ = wsf + WS_BK;
    float* qn  = wsf + WS_QN;
    float* kn  = wsf + WS_KN;

    hipLaunchKernelGGL(pack_weights, dim3(32), dim3(256), 0, stream, W1, W2, wp);
    hipLaunchKernelGGL(row_prep, dim3(B_ * NQ_), dim3(128), 0, stream,
                       q_inv, q_equi, Wq, bq, W1, 0, (const float*)nullptr, A, qn, DQI);
    hipLaunchKernelGGL(row_prep, dim3(B_ * NK_), dim3(128), 0, stream,
                       k_inv, k_equi, Wk, bk, W1, 64, b1, Bk_, kn, DKI);
    hipLaunchKernelGGL(pair_mlp, dim3(1024), dim3(128), 0, stream,
                       q_equi, k_equi, A, Bk_, qn, kn, wp, b2, out);
}
